// LeNet5_TTN3d_5978594476495
// MI455X (gfx1250) — matrix operand order repaired and checked by reading, not compiled, not run on hardware
//
#include <hip/hip_runtime.h>
#include <hip/hip_bf16.h>

typedef __attribute__((ext_vector_type(16))) _Float16 v16h;
typedef __attribute__((ext_vector_type(8)))  _Float16 v8h;
typedef __attribute__((ext_vector_type(8)))  float    v8f;

// ---------------------------------------------------------------------------
// Bilinear sample with zero padding (matches reference _grid_sample_2d corner)
// ---------------------------------------------------------------------------
__device__ static inline float bisample(const float* img, int H, int W,
                                        float iy, float ix) {
  float iy0f = floorf(iy), ix0f = floorf(ix);
  float wy = iy - iy0f, wx = ix - ix0f;
  int iy0 = (int)iy0f, ix0 = (int)ix0f;
  float acc = 0.f;
#pragma unroll
  for (int dy = 0; dy < 2; ++dy) {
#pragma unroll
    for (int dx = 0; dx < 2; ++dx) {
      int yy = iy0 + dy, xx = ix0 + dx;
      float wgt = (dy ? wy : 1.f - wy) * (dx ? wx : 1.f - wx);
      if (yy >= 0 && yy < H && xx >= 0 && xx < W)
        acc += img[yy * W + xx] * wgt;
    }
  }
  return acc;
}

__device__ static inline void affine_ixiy(float S, float R, float TX, float TY,
                                          int h, int w, float& iy, float& ix) {
  float c = S * cosf(R), s = S * sinf(R);
  float gxv = (2.f * w + 1.f) / 5.f - 1.f;
  float gyv = (2.f * h + 1.f) / 5.f - 1.f;
  float gx = c * gxv - s * gyv + (TX * c - TY * s);
  float gy = s * gxv + c * gyv + (TX * s + TY * c);
  ix = ((gx + 1.f) * 5.f - 1.f) * 0.5f;
  iy = ((gy + 1.f) * 5.f - 1.f) * 0.5f;
}

// ---------------------------------------------------------------------------
// Build weight1: (6,1,5,5,5) from fw1 (6,1,1,5,5) + 4 grid-sample steps
// ---------------------------------------------------------------------------
__global__ __launch_bounds__(256) void build_w1(const float* fw1, const float* sc,
                                                const float* rot, const float* tx,
                                                const float* ty, float* w1) {
  __shared__ float buf[2][6 * 25];
  int tid = threadIdx.x;
  if (tid < 150) {
    int o = tid / 25, p = tid % 25;
    float v = fw1[(o >> 1) * 25 + p];
    buf[0][o * 25 + p] = v;
    w1[o * 125 + p] = v;  // t = 0 slice
  }
  __syncthreads();
  for (int t = 1; t <= 4; ++t) {
    int cur = t & 1, prv = cur ^ 1;
    if (tid < 150) {
      int o = tid / 25, p = tid % 25, h = p / 5, w = p % 5;
      float iy, ix;
      affine_ixiy(sc[(t - 1) * 6 + o], rot[(t - 1) * 6 + o],
                  tx[(t - 1) * 6 + o], ty[(t - 1) * 6 + o], h, w, iy, ix);
      float v = bisample(&buf[prv][o * 25], 5, 5, iy, ix);
      buf[cur][o * 25 + p] = v;
      w1[o * 125 + t * 25 + p] = v;
    }
    __syncthreads();
  }
}

// ---------------------------------------------------------------------------
// Build weight2: (16,6,5,5,5) from fw2 (16,6,1,5,5) + 4 grid-sample steps
// ---------------------------------------------------------------------------
__global__ __launch_bounds__(256) void build_w2(const float* fw2, const float* sc,
                                                const float* rot, const float* tx,
                                                const float* ty, float* w2) {
  __shared__ float buf[2][16 * 6 * 25];
  int tid = threadIdx.x;
  for (int e = tid; e < 2400; e += 256) {
    int o = e / 150, r = e % 150, c = r / 25, p = r % 25;
    float v = fw2[((o >> 1) * 6 + c) * 25 + p];
    buf[0][e] = v;
    w2[o * 750 + c * 125 + p] = v;  // t = 0 slice
  }
  __syncthreads();
  for (int t = 1; t <= 4; ++t) {
    int cur = t & 1, prv = cur ^ 1;
    for (int e = tid; e < 2400; e += 256) {
      int o = e / 150, r = e % 150, c = r / 25, p = r % 25;
      int h = p / 5, w = p % 5;
      float iy, ix;
      affine_ixiy(sc[(t - 1) * 16 + o], rot[(t - 1) * 16 + o],
                  tx[(t - 1) * 16 + o], ty[(t - 1) * 16 + o], h, w, iy, ix);
      float v = bisample(&buf[prv][(o * 6 + c) * 25], 5, 5, iy, ix);
      buf[cur][e] = v;
      w2[o * 750 + c * 125 + t * 25 + p] = v;
    }
    __syncthreads();
  }
}

// ---------------------------------------------------------------------------
// Pack FC weights into zero-padded f16, N-major (n * Kpad + k) so a WMMA
// B-fragment load is two contiguous b128 loads per lane.
//   W1h: 128 x 2048  (from fc1_w 120 x 2000)
//   W2h:  96 x  128  (from fc2_w  84 x  120)
//   W3h:  16 x   96  (from fc3_w  10 x   84)
// ---------------------------------------------------------------------------
__global__ __launch_bounds__(256) void pack_weights(const float* __restrict__ fc1_w,
                                                    const float* __restrict__ fc2_w,
                                                    const float* __restrict__ fc3_w,
                                                    _Float16* __restrict__ W1h,
                                                    _Float16* __restrict__ W2h,
                                                    _Float16* __restrict__ W3h) {
  int tid = blockIdx.x * blockDim.x + threadIdx.x;
  int stride = gridDim.x * blockDim.x;
  for (int i = tid; i < 128 * 2048; i += stride) {
    int n = i >> 11, k = i & 2047;
    W1h[i] = (n < 120 && k < 2000) ? (_Float16)fc1_w[n * 2000 + k] : (_Float16)0.f;
  }
  for (int i = tid; i < 96 * 128; i += stride) {
    int n = i >> 7, k = i & 127;
    W2h[i] = (n < 84 && k < 120) ? (_Float16)fc2_w[n * 120 + k] : (_Float16)0.f;
  }
  for (int i = tid; i < 16 * 96; i += stride) {
    int n = i / 96, k = i % 96;
    W3h[i] = (n < 10 && k < 84) ? (_Float16)fc3_w[n * 84 + k] : (_Float16)0.f;
  }
}

// ---------------------------------------------------------------------------
// Conv1 (pad=2, k=5^3, Cin=1) + bias + relu + maxpool2 fused.
// Block = (pooled z, batch n); 224 threads, 196 active for the 14x14 plane.
// pool1 layout: [n][oc][dz][ph*14+pw]
// ---------------------------------------------------------------------------
__global__ __launch_bounds__(224) void conv1_pool(const float* __restrict__ x,
                                                  const float* __restrict__ w1,
                                                  const float* __restrict__ b1,
                                                  float* __restrict__ pool1) {
  __shared__ float sl[6 * 784];  // 6 input depth planes of 28x28
  __shared__ float ws[750];
  int dz = blockIdx.x, n = blockIdx.y;
  int tid = threadIdx.x;
  for (int i = tid; i < 750; i += 224) ws[i] = w1[i];
  const float* xn = x + (size_t)n * 28 * 28 * 28;
  for (int i = tid; i < 6 * 784; i += 224) {
    int slz = i / 784, p = i % 784;
    int z = 2 * dz - 2 + slz;
    float v = 0.f;
    if (z >= 0 && z < 28) {
      __builtin_prefetch(xn + z * 784 + p + 784, 0, 1);
      v = xn[z * 784 + p];
    }
    sl[i] = v;
  }
  __syncthreads();
  if (tid < 196) {
    int ph = tid / 14, pw = tid % 14;
    for (int oc = 0; oc < 6; ++oc) {
      float bias = b1[oc];
      float m = -3.4e38f;
#pragma unroll
      for (int dd = 0; dd < 2; ++dd) {
#pragma unroll
        for (int hh = 0; hh < 2; ++hh) {
#pragma unroll
          for (int ww = 0; ww < 2; ++ww) {
            int ch = 2 * ph + hh, cw = 2 * pw + ww;
            float acc = 0.f;
            for (int t = 0; t < 5; ++t) {
              const float* pl = &sl[(dd + t) * 784];
              const float* wt = &ws[oc * 125 + t * 25];
              for (int kh = 0; kh < 5; ++kh) {
                int ih = ch - 2 + kh;
                if (ih < 0 || ih >= 28) continue;
#pragma unroll
                for (int kw = 0; kw < 5; ++kw) {
                  int iw = cw - 2 + kw;
                  if (iw >= 0 && iw < 28)
                    acc += pl[ih * 28 + iw] * wt[kh * 5 + kw];
                }
              }
            }
            float v = acc + bias;
            v = v > 0.f ? v : 0.f;
            m = fmaxf(m, v);
          }
        }
      }
      pool1[(((size_t)n * 6 + oc) * 14 + dz) * 196 + tid] = m;
    }
  }
}

// ---------------------------------------------------------------------------
// Conv2 (pad=0, k=5^3, Cin=6) + bias + relu + maxpool2 fused; writes the
// flattened activation matrix A (256 x 2048, f16, zero-padded K) for WMMA FC.
// Block = (pooled z pd in 0..4, batch n); 128 threads.
// ---------------------------------------------------------------------------
__global__ __launch_bounds__(128) void conv2_pool(const float* __restrict__ pool1,
                                                  const float* __restrict__ w2,
                                                  const float* __restrict__ b2,
                                                  _Float16* __restrict__ A) {
  __shared__ float sl[6 * 6 * 196];  // 6 channels x 6 depth planes x 14x14
  int pd = blockIdx.x, n = blockIdx.y;
  int tid = threadIdx.x;
  for (int i = tid; i < 6 * 6 * 196; i += 128) {
    int ci = i / (6 * 196);
    int r = i % (6 * 196);
    int dl = r / 196, p = r % 196;
    sl[i] = pool1[(((size_t)n * 6 + ci) * 14 + (2 * pd + dl)) * 196 + p];
  }
  __syncthreads();
  for (int idx = tid; idx < 400; idx += 128) {
    int oc = idx / 25, rr = idx % 25, ph = rr / 5, pw = rr % 5;
    float bias = b2[oc];
    float m = -3.4e38f;
#pragma unroll
    for (int dd = 0; dd < 2; ++dd) {
#pragma unroll
      for (int hh = 0; hh < 2; ++hh) {
#pragma unroll
        for (int ww = 0; ww < 2; ++ww) {
          int h0 = 2 * ph + hh, w0 = 2 * pw + ww;
          float acc = 0.f;
          for (int ci = 0; ci < 6; ++ci) {
            const float* base = &sl[ci * 6 * 196];
            const float* wt = &w2[oc * 750 + ci * 125];
            for (int t = 0; t < 5; ++t) {
              for (int kh = 0; kh < 5; ++kh) {
                const float* row = &base[(dd + t) * 196 + (h0 + kh) * 14 + w0];
                const float* wr = &wt[t * 25 + kh * 5];
#pragma unroll
                for (int kw = 0; kw < 5; ++kw) acc += row[kw] * wr[kw];
              }
            }
          }
          float v = acc + bias;
          v = v > 0.f ? v : 0.f;
          m = fmaxf(m, v);
        }
      }
    }
    A[(size_t)n * 2048 + oc * 125 + pd * 25 + rr] = (_Float16)m;
  }
  if (pd == 0 && tid < 48) A[(size_t)n * 2048 + 2000 + tid] = (_Float16)0.f;
}

// ---------------------------------------------------------------------------
// WMMA fragment loaders (ISA 7.12.2 layouts, wave32). All loads are
// unconditional contiguous 16-byte vector loads.
// ---------------------------------------------------------------------------
__device__ static inline v16h cat16(v8h lo, v8h hi) {
  return __builtin_shufflevector(lo, hi, 0, 1, 2, 3, 4, 5, 6, 7,
                                 8, 9, 10, 11, 12, 13, 14, 15);
}

// A: 16x32 f16 tile at (row0, k0); caller passes Ap = base + row0*lda + k0.
__device__ static inline v16h load_A_f16(const _Float16* Ap, int lda, int lane) {
  int r = lane & 15, hi = lane >> 4;
  const _Float16* row = Ap + (size_t)r * lda + hi * 8;
  v8h c0 = *(const v8h*)(row);
  v8h c1 = *(const v8h*)(row + 16);
  return cat16(c0, c1);
}

// B: 32x16 tile of packed W (N-major, n*Kpad + k, zero-padded f16).
__device__ static inline v16h load_B_f16(const _Float16* Wp, int Kpad,
                                         int n0, int k0, int lane) {
  int c = lane & 15, koff = (lane >> 4) * 8;
  const _Float16* row = Wp + (size_t)(n0 + c) * Kpad + k0 + koff;
  v8h c0 = *(const v8h*)(row);
  v8h c1 = *(const v8h*)(row + 16);
  return cat16(c0, c1);
}

// ---------------------------------------------------------------------------
// Fused FC stack: relu(A@W1^T+b1) -> relu(@W2^T+b2) -> @W3^T+b3
// One block per 16-row M tile; 8 waves. fc1: each wave owns a 16-col N chunk.
// Intermediates kept in LDS as f16 (zero-padded), WMMA f32 accumulation.
// ---------------------------------------------------------------------------
__global__ __launch_bounds__(256) void fc_fused(const _Float16* __restrict__ A,
                                                const _Float16* __restrict__ W1h,
                                                const float* __restrict__ B1,
                                                const _Float16* __restrict__ W2h,
                                                const float* __restrict__ B2,
                                                const _Float16* __restrict__ W3h,
                                                const float* __restrict__ B3,
                                                float* __restrict__ out) {
  __shared__ __align__(16) _Float16 h1[16 * 128];  // fc1 out, cols >=120 zero
  __shared__ __align__(16) _Float16 h2[16 * 96];   // fc2 out, cols >=84 zero
  int m0 = blockIdx.x * 16;
  int wave = threadIdx.x >> 5;
  int lane = threadIdx.x & 31;
  int cl = lane & 15;          // column within 16-wide N chunk
  int mrow = (lane >> 4) * 8;  // 0 or 8: row-half of the C/D tile

  // ---- fc1: K = 2048 (padded), N chunks of 16, one per wave -------------
  {
    int n0 = wave * 16;
    v8f acc = {};
    const _Float16* Abase = A + (size_t)m0 * 2048;
    for (int k0 = 0; k0 < 2048; k0 += 32) {
      v16h a = load_A_f16(Abase + k0, 2048, lane);
      v16h b = load_B_f16(W1h, 2048, n0, k0, lane);
      acc = __builtin_amdgcn_wmma_f32_16x16x32_f16(false, a, false, b,
                                                   (short)0, acc, false, false);
    }
    int col = n0 + cl;
    float bias = (col < 120) ? B1[col] : 0.f;
#pragma unroll
    for (int i = 0; i < 8; ++i) {
      float v = acc[i] + bias;
      v = v > 0.f ? v : 0.f;
      h1[(mrow + i) * 128 + col] = (_Float16)v;
    }
  }
  __syncthreads();

  // ---- fc2: K = 128 (padded), N = 96 (padded) -> waves 0..5 -------------
  if (wave < 6) {
    int n0 = wave * 16;
    v8f acc = {};
#pragma unroll
    for (int k0 = 0; k0 < 128; k0 += 32) {
      v16h a = load_A_f16(&h1[k0], 128, lane);
      v16h b = load_B_f16(W2h, 128, n0, k0, lane);
      acc = __builtin_amdgcn_wmma_f32_16x16x32_f16(false, a, false, b,
                                                   (short)0, acc, false, false);
    }
    int col = n0 + cl;
    float bias = (col < 84) ? B2[col] : 0.f;
#pragma unroll
    for (int i = 0; i < 8; ++i) {
      float v = acc[i] + bias;
      v = v > 0.f ? v : 0.f;
      h2[(mrow + i) * 96 + col] = (_Float16)v;
    }
  }
  __syncthreads();

  // ---- fc3: K = 96 (padded), N = 16 (10 valid) -> wave 0 ----------------
  if (wave == 0) {
    v8f acc = {};
#pragma unroll
    for (int k0 = 0; k0 < 96; k0 += 32) {
      v16h a = load_A_f16(&h2[k0], 96, lane);
      v16h b = load_B_f16(W3h, 96, 0, k0, lane);
      acc = __builtin_amdgcn_wmma_f32_16x16x32_f16(false, a, false, b,
                                                   (short)0, acc, false, false);
    }
    if (cl < 10) {
      float bias = B3[cl];
#pragma unroll
      for (int i = 0; i < 8; ++i)
        out[(size_t)(m0 + mrow + i) * 10 + cl] = acc[i] + bias;
    }
  }
}

// ---------------------------------------------------------------------------
extern "C" void kernel_launch(void* const* d_in, const int* in_sizes, int n_in,
                              void* d_out, int out_size, void* d_ws, size_t ws_size,
                              hipStream_t stream) {
  const float* x     = (const float*)d_in[0];
  const float* fw1   = (const float*)d_in[1];
  const float* sc1   = (const float*)d_in[2];
  const float* rot1  = (const float*)d_in[3];
  const float* tx1   = (const float*)d_in[4];
  const float* ty1   = (const float*)d_in[5];
  const float* b1    = (const float*)d_in[6];
  const float* fw2   = (const float*)d_in[7];
  const float* sc2   = (const float*)d_in[8];
  const float* rot2  = (const float*)d_in[9];
  const float* tx2   = (const float*)d_in[10];
  const float* ty2   = (const float*)d_in[11];
  const float* b2    = (const float*)d_in[12];
  const float* fc1_w = (const float*)d_in[13];
  const float* fc1_b = (const float*)d_in[14];
  const float* fc2_w = (const float*)d_in[15];
  const float* fc2_b = (const float*)d_in[16];
  const float* fc3_w = (const float*)d_in[17];
  const float* fc3_b = (const float*)d_in[18];

  float* ws    = (float*)d_ws;
  float* w1    = ws;               // 750 floats
  float* w2    = ws + 1024;        // 12000 floats
  float* pool1 = ws + 16384;       // 256*6*14*14*14 = 4,214,784 floats
  _Float16* hreg = (_Float16*)(ws + 16384 + 4214784);
  _Float16* Am  = hreg;             // 256*2048   = 524288 halves
  _Float16* W1h = hreg + 524288;    // 128*2048   = 262144 halves
  _Float16* W2h = W1h + 262144;     // 96*128     =  12288 halves
  _Float16* W3h = W2h + 12288;      // 16*96      =   1536 halves

  build_w1<<<1, 256, 0, stream>>>(fw1, sc1, rot1, tx1, ty1, w1);
  build_w2<<<1, 256, 0, stream>>>(fw2, sc2, rot2, tx2, ty2, w2);
  pack_weights<<<64, 256, 0, stream>>>(fc1_w, fc2_w, fc3_w, W1h, W2h, W3h);
  conv1_pool<<<dim3(14, 256), 224, 0, stream>>>(x, w1, b1, pool1);
  conv2_pool<<<dim3(5, 256), 128, 0, stream>>>(pool1, w2, b2, Am);
  fc_fused<<<16, 256, 0, stream>>>(Am, W1h, fc1_b, W2h, fc2_b,
                                   W3h, fc3_b, (float*)d_out);
}
